// ContrastiveLoss_76845554860147
// MI455X (gfx1250) — compile-verified, matrix operand order checked by reading, not run on hardware
//
#include <hip/hip_runtime.h>
#include <hip/hip_bf16.h>
#include <stdint.h>

typedef __attribute__((ext_vector_type(16))) _Float16 v16h;
typedef __attribute__((ext_vector_type(8)))  _Float16 v8h;
typedef __attribute__((ext_vector_type(4)))  _Float16 v4h;
typedef __attribute__((ext_vector_type(8)))  float    v8f;
typedef __attribute__((ext_vector_type(4)))  float    v4f;

#define N_ROWS 8192
#define DIM    1024
#define INV_T  14.285714285714286f   // 1 / 0.07

// Load one 16-half WMMA fragment (A-row or B-col) for this lane.
// Per §7.12.2 16-bit layout: lane half (lane>>4) selects K sub-offset 0/8;
// elements 0..7 come from k+kofs, elements 8..15 from k+16+kofs.
__device__ __forceinline__ v16h load_frag(const _Float16* p) {
  v8h lo = *(const v8h*)(p);
  v8h hi = *(const v8h*)(p + 16);
  return __builtin_shufflevector(lo, hi,
           0, 1, 2, 3, 4, 5, 6, 7, 8, 9, 10, 11, 12, 13, 14, 15);
}

__device__ __forceinline__ v8f wmma_f16(v16h a, v16h b, v8f c) {
  return __builtin_amdgcn_wmma_f32_16x16x32_f16(
      /*neg_a=*/false, a, /*neg_b=*/false, b,
      /*c_mod=*/(short)0, c, /*reuse_a=*/false, /*reuse_b=*/false);
}

// ---------------------------------------------------------------------------
// Phase 0: zero the global loss accumulator (d_ws is poisoned by the harness)
// ---------------------------------------------------------------------------
__global__ void ContrastiveLoss_zero_kernel(float* acc) { *acc = 0.0f; }

// ---------------------------------------------------------------------------
// Phase 1: L2-normalize each row (f32 in) -> f16 workspace copy (row-major)
// ---------------------------------------------------------------------------
__global__ void __launch_bounds__(256)
ContrastiveLoss_normalize_kernel(const float* __restrict__ emb,
                                 _Float16* __restrict__ out) {
  const int row = blockIdx.x;
  const int tid = threadIdx.x;
  const float* rp = emb + (size_t)row * DIM;

  v4f x = *(const v4f*)(rp + tid * 4);
  float ss = x.x * x.x + x.y * x.y + x.z * x.z + x.w * x.w;
#pragma unroll
  for (int m = 1; m < 32; m <<= 1) ss += __shfl_xor(ss, m, 32);

  __shared__ float s_part[8];
  __shared__ float s_inv;
  if ((tid & 31) == 0) s_part[tid >> 5] = ss;
  __syncthreads();
  if (tid == 0) {
    float t = 0.0f;
#pragma unroll
    for (int i = 0; i < 8; ++i) t += s_part[i];
    float n = sqrtf(t);
    s_inv = 1.0f / fmaxf(n, 1e-12f);
  }
  __syncthreads();
  const float inv = s_inv;

  v4h h;
  h.x = (_Float16)(x.x * inv);
  h.y = (_Float16)(x.y * inv);
  h.z = (_Float16)(x.z * inv);
  h.w = (_Float16)(x.w * inv);
  *(v4h*)(out + (size_t)row * DIM + tid * 4) = h;
}

// ---------------------------------------------------------------------------
// Phase 2: fused  S = E*E^T / T -> exp -> masked row sums -> per-row loss.
// Block = 8 waves, 32-row A band staged in LDS (64 KB) via the CDNA5 async
// Global->LDS path (ASYNCcnt). Each wave register-blocks a 2x4 grid of 16x16
// WMMA tiles (32 rows x 64 cols): per K-step, 2 A frags (4 ds loads) + 4 B
// frags (8 global loads) feed 8 WMMAs -> 1.5 fragment loads per WMMA.
// ---------------------------------------------------------------------------
__global__ void __launch_bounds__(256)
ContrastiveLoss_supcon_kernel(const _Float16* __restrict__ Ef16,
                              const int* __restrict__ labels,
                              float* __restrict__ acc) {
  __shared__ _Float16 s_A[32 * DIM];   // 64 KB: 32-row A band
  __shared__ float s_rowAll[32];
  __shared__ float s_rowPos[32];

  const int tid  = threadIdx.x;
  const int wave = tid >> 5;
  const int lane = tid & 31;
  const int half = lane >> 4;          // selects K sub-offset (wave32 layout)
  const int l15  = lane & 15;
  const int m0   = blockIdx.x * 32;

  // Stage the 32-row A band into LDS with async global->LDS b128 transfers
  // (no VGPR round-trip; tracked by ASYNCcnt).
  {
    const unsigned lds0 = (unsigned)(uintptr_t)s_A;   // low 32 bits = LDS offset
    const char* g0 = (const char*)(Ef16 + (size_t)m0 * DIM);
#pragma unroll
    for (int it = 0; it < (32 * DIM) / 8 / 256; ++it) {
      const int t = it * 256 + tid;
      asm volatile("global_load_async_to_lds_b128 %0, %1, off"
                   :: "v"(lds0 + (unsigned)t * 16u),
                      "v"(g0 + (size_t)t * 16)
                   : "memory");
    }
    asm volatile("s_wait_asynccnt 0x0" ::: "memory");
  }
  if (tid < 32) { s_rowAll[tid] = 0.0f; s_rowPos[tid] = 0.0f; }
  __syncthreads();

  // Labels of the 16 rows this lane contributes to (2 bands x 8 C-VGPRs).
  int labRow[2][8];
#pragma unroll
  for (int b = 0; b < 2; ++b)
#pragma unroll
    for (int r = 0; r < 8; ++r)
      labRow[b][r] = labels[m0 + b * 16 + r + 8 * half];

  const int kofs = half * 8;
  const _Float16* aRow0 = s_A + (size_t)l15 * DIM + kofs;
  const _Float16* aRow1 = s_A + (size_t)(16 + l15) * DIM + kofs;

  float accAll[2][8], accPos[2][8];
#pragma unroll
  for (int b = 0; b < 2; ++b)
#pragma unroll
    for (int r = 0; r < 8; ++r) { accAll[b][r] = 0.0f; accPos[b][r] = 0.0f; }

  for (int q = wave; q < N_ROWS / 64; q += 8) {
    const int n0 = q * 64;
    int colg[4], labC[4];
    const _Float16* brow[4];
#pragma unroll
    for (int t = 0; t < 4; ++t) {
      colg[t] = n0 + t * 16 + l15;                       // column this lane owns
      brow[t] = Ef16 + (size_t)colg[t] * DIM + kofs;     // B = E^T
      labC[t] = labels[colg[t]];
      // Prefetch next quad's B rows.
      if (q + 8 < N_ROWS / 64)
        __builtin_prefetch(Ef16 + (size_t)(colg[t] + 8 * 64) * DIM, 0, 0);
    }

    v8f c[2][4];
#pragma unroll
    for (int t = 0; t < 4; ++t) { c[0][t] = (v8f){}; c[1][t] = (v8f){}; }

#pragma unroll 1
    for (int k = 0; k < DIM; k += 32) {
      const v16h A0 = load_frag(aRow0 + k);   // ds_load_b128 x2
      const v16h A1 = load_frag(aRow1 + k);   // ds_load_b128 x2
#pragma unroll
      for (int t = 0; t < 4; ++t) {
        const v16h B = load_frag(brow[t] + k); // global_load_b128 x2
        c[0][t] = wmma_f16(A0, B, c[0][t]);
        c[1][t] = wmma_f16(A1, B, c[1][t]);
      }
    }

    // Fused epilogue: exp + label/diagonal masking + per-lane accumulation.
#pragma unroll
    for (int t = 0; t < 4; ++t) {
#pragma unroll
      for (int r = 0; r < 8; ++r) {
        const int row0 = m0 + r + 8 * half;   // band 0 row
        const int row1 = row0 + 16;           // band 1 row
        const float e0 = __expf(c[0][t][r] * INV_T);
        accAll[0][r] += e0;
        accPos[0][r] += ((labC[t] == labRow[0][r]) && (colg[t] != row0)) ? e0 : 0.0f;
        const float e1 = __expf(c[1][t][r] * INV_T);
        accAll[1][r] += e1;
        accPos[1][r] += ((labC[t] == labRow[1][r]) && (colg[t] != row1)) ? e1 : 0.0f;
      }
    }
  }

  // Reduce across the 16 lanes sharing a row (wave32: width-16 segments),
  // then combine the 8 waves through LDS float atomics.
#pragma unroll
  for (int b = 0; b < 2; ++b)
#pragma unroll
    for (int r = 0; r < 8; ++r) {
      float a = accAll[b][r], pz = accPos[b][r];
#pragma unroll
      for (int m = 1; m < 16; m <<= 1) {
        a  += __shfl_xor(a,  m, 16);
        pz += __shfl_xor(pz, m, 16);
      }
      if (l15 == 0) {
        atomicAdd(&s_rowAll[b * 16 + r + 8 * half], a);
        atomicAdd(&s_rowPos[b * 16 + r + 8 * half], pz);
      }
    }
  __syncthreads();

  // loss_row = log(sumAll) - log(sumPos); block-sum -> one global atomic
  if (tid < 32) {
    float loss = __logf(s_rowAll[tid]) - __logf(s_rowPos[tid]);
#pragma unroll
    for (int m = 1; m < 32; m <<= 1) loss += __shfl_xor(loss, m, 32);
    if (tid == 0) atomicAdd(acc, loss);
  }
}

// ---------------------------------------------------------------------------
// Phase 3: mean
// ---------------------------------------------------------------------------
__global__ void ContrastiveLoss_finalize_kernel(const float* acc, float* out) {
  *out = *acc * (1.0f / (float)N_ROWS);
}

extern "C" void kernel_launch(void* const* d_in, const int* in_sizes, int n_in,
                              void* d_out, int out_size, void* d_ws, size_t ws_size,
                              hipStream_t stream) {
  const float* emb    = (const float*)d_in[0];
  const int*   labels = (const int*)d_in[1];
  float*       out    = (float*)d_out;

  _Float16* Ef16 = (_Float16*)d_ws;                                 // 16 MB
  float*    acc  = (float*)((char*)d_ws +
                            (size_t)N_ROWS * DIM * sizeof(_Float16));

  ContrastiveLoss_zero_kernel<<<1, 1, 0, stream>>>(acc);
  ContrastiveLoss_normalize_kernel<<<N_ROWS, 256, 0, stream>>>(emb, Ef16);
  ContrastiveLoss_supcon_kernel<<<N_ROWS / 32, 256, 0, stream>>>(Ef16, labels, acc);
  ContrastiveLoss_finalize_kernel<<<1, 1, 0, stream>>>(acc, out);
}
